// simulation_grid_model_v2_44590350467726
// MI455X (gfx1250) — matrix operand order repaired and checked by reading, not compiled, not run on hardware
//
#include <hip/hip_runtime.h>
#include <hip/hip_bf16.h>
#include <math.h>

typedef __attribute__((ext_vector_type(16))) _Float16 v16h;
typedef __attribute__((ext_vector_type(8)))  float    v8f;

// ---------------- problem constants ----------------
constexpr int CB   = 4;      // batch
constexpr int CS   = 500;    // stations
constexpr int CG   = 2304;   // grid cells (48*48)
constexpr int CTIN = 7;
constexpr int CTOUT= 6;
constexpr int CF   = 12;
constexpr int CH   = 128;
constexpr int SIMSTEP = 28;
constexpr int NTOK = CG + CS;          // 2804
constexpr int NPAD = 2816;             // 176 * 16
constexpr int RS   = CB * CS;          // 2000 station rows
constexpr int RG   = CB * CG;          // 9216 grid rows
constexpr int RX   = CB * NPAD;        // 11264 token rows
constexpr int KENC = 192;              // 52 + 128 padded
constexpr int KDEC = 192;              // 40 + 128 padded
constexpr int KGRD = 224;              // 96 + 128
constexpr float PMM = 25.0f, PMS = 15.0f;
constexpr float ATT_SCALE = 0.0883883476483184f; // 1/sqrt(128)

// ---------------- parameter block ----------------
struct P {
  // inputs
  const float *feats, *raw_times, *simulation, *lats, *lons, *coords, *te_w, *te_b;
  const unsigned char *masks;
  const float *enc_wih,*enc_whh,*enc_bih,*enc_bhh;
  const float *dec_wih,*dec_whh,*dec_bih,*dec_bhh;
  const float *grid_wih,*grid_whh,*grid_bih,*grid_bhh;
  const float *eiw,*eib,*eow,*eob,*diw,*dib,*dow,*dob;
  const float *last_w,*last_b,*stn_h0,*stn_c0,*grid_h0,*grid_c0;
  // workspace
  float *tfe,*stn_loc,*grid_loc;
  float *stn_h,*stn_c,*grid_h,*grid_c;
  float *xh;
  _Float16 *xh_h, *qb, *kb, *vt, *ctx;
  _Float16 *Xs, *Xg;
  _Float16 *Wenc,*Wdec,*Wgrd,*Weiw,*Weow,*Wdiw,*Wdow;
  float *Benc,*Bdec,*Bgrd;
  float *out;
};

// ---------------- WMMA helpers ----------------
__device__ __forceinline__ v8f wmma16(v16h a, v16h b, v8f c) {
  return __builtin_amdgcn_wmma_f32_16x16x32_f16(false, a, false, b, (short)0, c,
                                                false, false);
}

// A fragment: rows rowBase+lane%16, K = kBase..kBase+31.
// halves 0..7 -> K = kBase + 8*grp + h ; halves 8..15 -> K = kBase + 16 + 8*grp + (h-8)
__device__ __forceinline__ v16h ldfragA(const _Float16* mat, int ld, int rowBase, int kBase) {
  int lane = threadIdx.x & 31;
  int grp  = (lane >> 4) & 1;
  const _Float16* pp = mat + (size_t)(rowBase + (lane & 15)) * ld + kBase + (grp << 3);
  union { v16h v; float4 q[2]; } u;
  u.q[0] = *(const float4*)(pp);
  u.q[1] = *(const float4*)(pp + 16);
  return u.v;
}

// B fragment: cols nBase+lane%16 (stored as rows of mat), same K layout as A (kBase + 8*grp + h ; +16 for halves 8..15)
__device__ __forceinline__ v16h ldfragB(const _Float16* mat, int ld, int nBase, int kBase) {
  int lane = threadIdx.x & 31;
  int grp  = (lane >> 4) & 1;
  const _Float16* pp = mat + (size_t)(nBase + (lane & 15)) * ld + kBase + (grp << 3);
  union { v16h v; float4 q[2]; } u;
  u.q[0] = *(const float4*)(pp);
  u.q[1] = *(const float4*)(pp + 16);
  return u.v;
}

__device__ __forceinline__ v8f zero8() {
  return v8f{0.f,0.f,0.f,0.f,0.f,0.f,0.f,0.f};
}
__device__ __forceinline__ float sigmoidf_(float x) { return 1.f / (1.f + __expf(-x)); }

// ---------------- weight prep (fp32 -> f16, concat [Wih|Whh], combined bias) --------
__global__ void prep_weights_kernel(P p) {
  const int nEnc = 512*KENC, nDec = 512*KDEC, nGrd = 512*KGRD;
  const int nIn = 384*128, nOut = 128*128;
  int t = blockIdx.x * 256 + threadIdx.x;
  if (t < nEnc) { int n=t/KENC,k=t%KENC;
    float v = (k<52)? p.enc_wih[n*52+k] : (k<180 ? p.enc_whh[n*128+(k-52)] : 0.f);
    p.Wenc[t]=(_Float16)v; return; }
  t -= nEnc;
  if (t < nDec) { int n=t/KDEC,k=t%KDEC;
    float v = (k<40)? p.dec_wih[n*40+k] : (k<168 ? p.dec_whh[n*128+(k-40)] : 0.f);
    p.Wdec[t]=(_Float16)v; return; }
  t -= nDec;
  if (t < nGrd) { int n=t/KGRD,k=t%KGRD;
    float v = (k<96)? p.grid_wih[n*96+k] : p.grid_whh[n*128+(k-96)];
    p.Wgrd[t]=(_Float16)v; return; }
  t -= nGrd;
  if (t < nIn)  { p.Weiw[t]=(_Float16)p.eiw[t]; return; }  t -= nIn;
  if (t < nOut) { p.Weow[t]=(_Float16)p.eow[t]; return; }  t -= nOut;
  if (t < nIn)  { p.Wdiw[t]=(_Float16)p.diw[t]; return; }  t -= nIn;
  if (t < nOut) { p.Wdow[t]=(_Float16)p.dow[t]; return; }  t -= nOut;
  if (t < 512)  { p.Benc[t]=p.enc_bih[t]+p.enc_bhh[t]; return; } t -= 512;
  if (t < 512)  { p.Bdec[t]=p.dec_bih[t]+p.dec_bhh[t]; return; } t -= 512;
  if (t < 512)  { p.Bgrd[t]=p.grid_bih[t]+p.grid_bhh[t]; return; }
}

// ---------------- precompute tfe / stn_loc / grid_loc ----------------
__global__ void precompute_kernel(P p) {
  const int nT = CB*13*24, nS = CS*16, nG = CG*16;
  int t = blockIdx.x*256 + threadIdx.x;
  if (t < nT) {
    int b = t/(13*24); int r = t%(13*24); int ti = r/24; int k = r%24;
    int j = k/8, u = k%8, tt = u & 3;
    float ph = p.raw_times[(b*13+ti)*3 + j] * p.te_w[(2+j)*4+tt] + p.te_b[(2+j)*4+tt];
    p.tfe[t] = (u<4) ? __sinf(ph) : __cosf(ph);
    return;
  }
  t -= nT;
  if (t < nS) {
    int s = t/16, k = t%16, half = k/8, u = k%8, tt = u & 3;
    float cv = half ? p.lons[s] : p.lats[s];
    float ph = cv * p.te_w[half*4+tt] + p.te_b[half*4+tt];
    p.stn_loc[t] = (u<4) ? __sinf(ph) : __cosf(ph);
    return;
  }
  t -= nS;
  if (t < nG) {
    int g = t/16, k = t%16, half = k/8, u = k%8, tt = u & 3;
    float cv = p.coords[g*2 + half];
    float ph = cv * p.te_w[half*4+tt] + p.te_b[half*4+tt];
    p.grid_loc[t] = (u<4) ? __sinf(ph) : __cosf(ph);
  }
}

// ---------------- init hidden/cell state ----------------
__global__ void init_state_kernel(P p) {
  const int nS = RS*CH, nG = RG*CH;
  int idx = blockIdx.x*256 + threadIdx.x;
  if (idx < nS) {
    int src = idx % (CS*CH);
    p.stn_h[idx] = p.stn_h0[src];
    p.stn_c[idx] = p.stn_c0[src];
  } else if (idx < nS + nG) {
    int t = idx - nS;
    int src = t % (CG*CH);
    p.grid_h[t] = p.grid_h0[src];
    p.grid_c[t] = p.grid_c0[src];
  }
}

// ---------------- LSTM input packing ----------------
__global__ void build_stn_enc_kernel(P p, int step) {
  int idx = blockIdx.x*256 + threadIdx.x;
  if (idx >= RS*KENC) return;
  int row = idx / KENC, k = idx % KENC;
  int b = row / CS, s = row % CS;
  float v;
  if (k < 12) {
    v = p.feats[((size_t)(b*CTIN+step)*CS + s)*CF + k];
    if (k == 0) v = (v - PMM) / PMS;
  } else if (k < 36)  v = p.tfe[(b*13+step)*24 + (k-12)];
  else   if (k < 52)  v = p.stn_loc[s*16 + (k-36)];
  else   if (k < 180) v = p.stn_h[(size_t)row*CH + (k-52)];
  else                v = 0.f;
  p.Xs[idx] = (_Float16)v;
}

__global__ void build_grid_enc_kernel(P p, int step) {
  int idx = blockIdx.x*256 + threadIdx.x;
  if (idx >= RG*KGRD) return;
  int row = idx / KGRD, k = idx % KGRD;
  int b = row / CG, g = row % CG;
  float v;
  if (k < 24) v = p.tfe[(b*13+step)*24 + k];
  else if (k < 48) {
    int t = k - 24;
    v = p.simulation[((size_t)(b*CG+g))*(CTIN*SIMSTEP) + step*SIMSTEP + t];
    if (t==4 || t==10 || t==16 || t==22) v = (v - PMM) / PMS;
  } else if (k < 80) {
    int u = k - 48, j = u/8, q = u%8, tt = q & 3;
    float lv = p.simulation[((size_t)(b*CG+g))*(CTIN*SIMSTEP) + step*SIMSTEP + 24 + j];
    float ph = lv * p.te_w[5*4+tt] + p.te_b[5*4+tt];
    v = (q<4) ? __sinf(ph) : __cosf(ph);
  } else if (k < 96)  v = p.grid_loc[g*16 + (k-80)];
  else                v = p.grid_h[(size_t)row*CH + (k-96)];
  p.Xg[idx] = (_Float16)v;
}

__global__ void build_stn_dec_kernel(P p, int step) {
  int idx = blockIdx.x*256 + threadIdx.x;
  if (idx >= RS*KDEC) return;
  int row = idx / KDEC, k = idx % KDEC;
  int b = row / CS, s = row % CS;
  float v;
  if (k < 24)        v = p.tfe[(b*13 + CTIN + step)*24 + k];
  else if (k < 40)   v = p.stn_loc[s*16 + (k-24)];
  else if (k < 168)  v = p.stn_h[(size_t)row*CH + (k-40)];
  else               v = 0.f;
  p.Xs[idx] = (_Float16)v;
}

// ---------------- fused LSTM step: gates GEMM (WMMA) + nonlinearity ----------------
// block = 256 threads (8 waves); each block does 16 rows x 512 gate cols.
// KPAD is a compile-time constant so the K loop fully unrolls with literal offsets,
// and B fragments are preloaded per chunk so loads overlap the WMMA pipe.
template <int KPAD>
__global__ void lstm_kernel(const _Float16* __restrict__ X,
                            const _Float16* __restrict__ W, const float* __restrict__ bias,
                            float* __restrict__ h, float* __restrict__ c) {
  __shared__ __align__(16) float g_lds[16*512];
  int wave = threadIdx.x >> 5;
  int lane = threadIdx.x & 31;
  int rowBase = blockIdx.x * 16;
  v8f acc[4];
  #pragma unroll
  for (int t = 0; t < 4; ++t) acc[t] = zero8();
  #pragma unroll
  for (int kc = 0; kc < KPAD; kc += 32) {
    v16h a = ldfragA(X, KPAD, rowBase, kc);
    v16h bf[4];
    #pragma unroll
    for (int t = 0; t < 4; ++t) bf[t] = ldfragB(W, KPAD, (wave*4 + t)*16, kc);
    #pragma unroll
    for (int t = 0; t < 4; ++t) acc[t] = wmma16(a, bf[t], acc[t]);
  }
  int g = lane >> 4, cl = lane & 15;
  #pragma unroll
  for (int t = 0; t < 4; ++t) {
    int col = (wave*4+t)*16 + cl;
    float bcol = bias[col];
    #pragma unroll
    for (int r = 0; r < 8; ++r)
      g_lds[(r + 8*g)*512 + col] = acc[t][r] + bcol;
  }
  __syncthreads();
  for (int e = threadIdx.x; e < 2048; e += 256) {
    int row = e >> 7, col = e & 127;
    float gi = g_lds[row*512 + col];
    float gf = g_lds[row*512 + 128 + col];
    float gg = g_lds[row*512 + 256 + col];
    float go = g_lds[row*512 + 384 + col];
    size_t off = (size_t)(rowBase + row)*CH + col;
    float cn = sigmoidf_(gf) * c[off] + sigmoidf_(gi) * tanhf(gg);
    c[off] = cn;
    h[off] = sigmoidf_(go) * tanhf(cn);
  }
}

// ---------------- assemble xh = [grid_h ; stn_h ; pad] (f32 + f16) ----------------
__global__ void assemble_kernel(P p) {
  int idx = blockIdx.x*256 + threadIdx.x;
  if (idx >= CB*NPAD*CH) return;
  int b = idx / (NPAD*CH);
  int rem = idx % (NPAD*CH);
  int tok = rem / CH, hh = rem % CH;
  float v;
  if (tok < CG)        v = p.grid_h[((size_t)b*CG + tok)*CH + hh];
  else if (tok < NTOK) v = p.stn_h[((size_t)b*CS + (tok-CG))*CH + hh];
  else                 v = 0.f;
  p.xh[idx] = v;
  p.xh_h[idx] = (_Float16)v;
}

__global__ void unpack_kernel(P p) {
  int idx = blockIdx.x*256 + threadIdx.x;
  if (idx >= CB*NTOK*CH) return;
  int b = idx / (NTOK*CH);
  int rem = idx % (NTOK*CH);
  int tok = rem / CH, hh = rem % CH;
  float v = p.xh[((size_t)b*NPAD + tok)*CH + hh];
  if (tok < CG) p.grid_h[((size_t)b*CG + tok)*CH + hh] = v;
  else          p.stn_h[((size_t)b*CS + (tok-CG))*CH + hh] = v;
}

// ---------------- QKV projection (WMMA): q,k row-major f16; v transposed ------------
__global__ void qkv_kernel(const _Float16* __restrict__ xh_h,
                           const _Float16* __restrict__ Win, const float* __restrict__ bin,
                           _Float16* __restrict__ qb, _Float16* __restrict__ kb,
                           _Float16* __restrict__ vt) {
  __shared__ __align__(16) float g_lds[16*384];
  int wave = threadIdx.x >> 5;
  int lane = threadIdx.x & 31;
  int rowBase = blockIdx.x * 16;
  v8f acc[3];
  #pragma unroll
  for (int t = 0; t < 3; ++t) acc[t] = zero8();
  #pragma unroll
  for (int kc = 0; kc < CH; kc += 32) {
    v16h a = ldfragA(xh_h, CH, rowBase, kc);
    v16h bf[3];
    #pragma unroll
    for (int t = 0; t < 3; ++t) bf[t] = ldfragB(Win, CH, (wave*3 + t)*16, kc);
    #pragma unroll
    for (int t = 0; t < 3; ++t) acc[t] = wmma16(a, bf[t], acc[t]);
  }
  int g = lane >> 4, cl = lane & 15;
  #pragma unroll
  for (int t = 0; t < 3; ++t) {
    int col = (wave*3+t)*16 + cl;
    float bcol = bin[col];
    #pragma unroll
    for (int r = 0; r < 8; ++r)
      g_lds[(r + 8*g)*384 + col] = acc[t][r] + bcol;
  }
  __syncthreads();
  for (int e = threadIdx.x; e < 16*384; e += 256) {
    int row = e / 384, col = e % 384;
    int R = rowBase + row;
    _Float16 v = (_Float16)g_lds[row*384 + col];
    if (col < 128)       qb[(size_t)R*CH + col] = v;
    else if (col < 256)  kb[(size_t)R*CH + (col-128)] = v;
    else {
      int b = R / NPAD, tok = R % NPAD;
      vt[((size_t)b*CH + (col-256))*NPAD + tok] = v;
    }
  }
}

// ---------------- flash attention: 1 wave per 16-query tile ----------------
__device__ __forceinline__ float keybias(const unsigned char* masks, int mode, int step,
                                         int b, int n) {
  if (n >= NTOK) return -1e30f;
  if (mode == 1) return 0.f;
  if (n < CG) return 0.f;
  return masks[(size_t)(b*CTIN + step)*CS + (n - CG)] ? 0.f : -1e30f;
}

__global__ void flash_kernel(const _Float16* __restrict__ qb, const _Float16* __restrict__ kb,
                             const _Float16* __restrict__ vt, _Float16* __restrict__ ctx,
                             const unsigned char* __restrict__ masks, int step, int mode) {
  __shared__ __align__(16) _Float16 p_lds[16*32];
  int lane = threadIdx.x;
  int b  = blockIdx.x / (NPAD/16);
  int qt = blockIdx.x % (NPAD/16);
  int rowBase = qt * 16;
  const _Float16* Q = qb + (size_t)b*NPAD*CH;
  const _Float16* K = kb + (size_t)b*NPAD*CH;
  const _Float16* V = vt + (size_t)b*CH*NPAD;
  int g = lane >> 4, cl = lane & 15;

  v16h qf[4];
  #pragma unroll
  for (int cch = 0; cch < 4; ++cch) qf[cch] = ldfragA(Q, CH, rowBase, cch*32);

  v8f acc[8];
  #pragma unroll
  for (int j = 0; j < 8; ++j) acc[j] = zero8();
  float mrun[8], lrun[8];
  #pragma unroll
  for (int r = 0; r < 8; ++r) { mrun[r] = -1e30f; lrun[r] = 0.f; }

  for (int kt = 0; kt < NPAD; kt += 32) {
    // ---- scores: preload all 8 K fragments, then 8 WMMAs ----
    v16h kf[8];
    #pragma unroll
    for (int cch = 0; cch < 4; ++cch) {
      kf[cch]     = ldfragB(K, CH, kt,      cch*32);
      kf[cch + 4] = ldfragB(K, CH, kt + 16, cch*32);
    }
    v8f s0 = zero8(), s1 = zero8();
    #pragma unroll
    for (int cch = 0; cch < 4; ++cch) {
      s0 = wmma16(qf[cch], kf[cch],     s0);
      s1 = wmma16(qf[cch], kf[cch + 4], s1);
    }
    float bias0 = keybias(masks, mode, step, b, kt + cl);
    float bias1 = keybias(masks, mode, step, b, kt + 16 + cl);
    #pragma unroll
    for (int r = 0; r < 8; ++r) {
      s0[r] = s0[r] * ATT_SCALE + bias0;
      s1[r] = s1[r] * ATT_SCALE + bias1;
    }
    #pragma unroll
    for (int r = 0; r < 8; ++r) {
      float t = fmaxf(s0[r], s1[r]);
      t = fmaxf(t, __shfl_xor(t, 1));
      t = fmaxf(t, __shfl_xor(t, 2));
      t = fmaxf(t, __shfl_xor(t, 4));
      t = fmaxf(t, __shfl_xor(t, 8));
      float mnew = fmaxf(mrun[r], t);
      float resc = __expf(mrun[r] - mnew);
      float p0 = __expf(s0[r] - mnew);
      float p1 = __expf(s1[r] - mnew);
      float ps = p0 + p1;
      ps += __shfl_xor(ps, 1);
      ps += __shfl_xor(ps, 2);
      ps += __shfl_xor(ps, 4);
      ps += __shfl_xor(ps, 8);
      lrun[r] = lrun[r] * resc + ps;
      mrun[r] = mnew;
      #pragma unroll
      for (int j = 0; j < 8; ++j) acc[j][r] *= resc;
      p_lds[(r + 8*g)*32 + cl]      = (_Float16)p0;
      p_lds[(r + 8*g)*32 + 16 + cl] = (_Float16)p1;
    }
    __syncthreads();
    // ---- P @ V: preload all 8 V fragments, then 8 WMMAs ----
    v16h pf = ldfragA(p_lds, 32, 0, 0);
    v16h vf[8];
    #pragma unroll
    for (int j = 0; j < 8; ++j) vf[j] = ldfragB(V, NPAD, j*16, kt);
    #pragma unroll
    for (int j = 0; j < 8; ++j) acc[j] = wmma16(pf, vf[j], acc[j]);
    __syncthreads();
  }
  _Float16* outp = ctx + ((size_t)b*NPAD + rowBase)*CH;
  #pragma unroll
  for (int r = 0; r < 8; ++r) {
    float inv = 1.f / fmaxf(lrun[r], 1e-20f);
    int row = r + 8*g;
    #pragma unroll
    for (int j = 0; j < 8; ++j)
      outp[(size_t)row*CH + j*16 + cl] = (_Float16)(acc[j][r] * inv);
  }
}

// ---------------- output projection + residual: xh += ctx @ Wout^T + b ----------------
__global__ void outproj_kernel(const _Float16* __restrict__ ctx,
                               const _Float16* __restrict__ Wout, const float* __restrict__ bout,
                               float* __restrict__ xh, _Float16* __restrict__ xh_h) {
  __shared__ __align__(16) float g_lds[16*128];
  int wave = threadIdx.x >> 5;
  int lane = threadIdx.x & 31;
  int rowBase = blockIdx.x * 16;
  v16h a[4], bf[4];
  #pragma unroll
  for (int kc = 0; kc < 4; ++kc) {
    a[kc]  = ldfragA(ctx, CH, rowBase, kc*32);
    bf[kc] = ldfragB(Wout, CH, wave*16, kc*32);
  }
  v8f acc = zero8();
  #pragma unroll
  for (int kc = 0; kc < 4; ++kc) acc = wmma16(a[kc], bf[kc], acc);
  int g = lane >> 4, cl = lane & 15;
  #pragma unroll
  for (int r = 0; r < 8; ++r)
    g_lds[(r + 8*g)*128 + wave*16 + cl] = acc[r];
  __syncthreads();
  for (int e = threadIdx.x; e < 2048; e += 256) {
    int row = e >> 7, col = e & 127;
    size_t off = (size_t)(rowBase + row)*CH + col;
    float nv = xh[off] + g_lds[row*128 + col] + bout[col];
    xh[off] = nv;
    xh_h[off] = (_Float16)nv;
  }
}

// ---------------- prediction head ----------------
__global__ void head_kernel(P p, int step) {
  int idx = blockIdx.x*256 + threadIdx.x;
  if (idx >= RS) return;
  int b = idx / CS, s = idx % CS;
  float acc = p.last_b[0];
  const float* hrow = p.stn_h + (size_t)idx*CH;
  #pragma unroll 8
  for (int hh = 0; hh < CH; ++hh) acc += hrow[hh] * p.last_w[hh];
  float v = acc * PMS + PMM;
  p.out[(size_t)(b*CTOUT + step)*CS + s] = fmaxf(v, 0.f);
}

// ---------------- host launcher ----------------
extern "C" void kernel_launch(void* const* d_in, const int* in_sizes, int n_in,
                              void* d_out, int out_size, void* d_ws, size_t ws_size,
                              hipStream_t stream) {
  (void)in_sizes; (void)n_in; (void)out_size; (void)ws_size;
  P p{};
  p.feats      = (const float*)d_in[0];
  p.masks      = (const unsigned char*)d_in[1];
  p.raw_times  = (const float*)d_in[2];
  p.simulation = (const float*)d_in[4];
  p.lats       = (const float*)d_in[5];
  p.lons       = (const float*)d_in[6];
  p.coords     = (const float*)d_in[7];
  p.te_w       = (const float*)d_in[8];
  p.te_b       = (const float*)d_in[9];
  p.enc_wih = (const float*)d_in[10]; p.enc_whh = (const float*)d_in[11];
  p.enc_bih = (const float*)d_in[12]; p.enc_bhh = (const float*)d_in[13];
  p.dec_wih = (const float*)d_in[14]; p.dec_whh = (const float*)d_in[15];
  p.dec_bih = (const float*)d_in[16]; p.dec_bhh = (const float*)d_in[17];
  p.grid_wih = (const float*)d_in[18]; p.grid_whh = (const float*)d_in[19];
  p.grid_bih = (const float*)d_in[20]; p.grid_bhh = (const float*)d_in[21];
  p.eiw = (const float*)d_in[22]; p.eib = (const float*)d_in[23];
  p.eow = (const float*)d_in[24]; p.eob = (const float*)d_in[25];
  p.diw = (const float*)d_in[26]; p.dib = (const float*)d_in[27];
  p.dow = (const float*)d_in[28]; p.dob = (const float*)d_in[29];
  p.last_w = (const float*)d_in[30]; p.last_b = (const float*)d_in[31];
  p.stn_h0 = (const float*)d_in[32]; p.stn_c0 = (const float*)d_in[33];
  p.grid_h0 = (const float*)d_in[34]; p.grid_c0 = (const float*)d_in[35];

  char* w = (char*)d_ws;
  auto alloc = [&](size_t bytes) -> char* {
    char* r = w; w += (bytes + 255) & ~(size_t)255; return r;
  };
  p.tfe      = (float*)alloc(CB*13*24*sizeof(float));
  p.stn_loc  = (float*)alloc(CS*16*sizeof(float));
  p.grid_loc = (float*)alloc(CG*16*sizeof(float));
  p.stn_h  = (float*)alloc((size_t)RS*CH*sizeof(float));
  p.stn_c  = (float*)alloc((size_t)RS*CH*sizeof(float));
  p.grid_h = (float*)alloc((size_t)RG*CH*sizeof(float));
  p.grid_c = (float*)alloc((size_t)RG*CH*sizeof(float));
  p.xh   = (float*)alloc((size_t)RX*CH*sizeof(float));
  p.xh_h = (_Float16*)alloc((size_t)RX*CH*sizeof(_Float16));
  p.qb   = (_Float16*)alloc((size_t)RX*CH*sizeof(_Float16));
  p.kb   = (_Float16*)alloc((size_t)RX*CH*sizeof(_Float16));
  p.vt   = (_Float16*)alloc((size_t)RX*CH*sizeof(_Float16));
  p.ctx  = (_Float16*)alloc((size_t)RX*CH*sizeof(_Float16));
  p.Xs   = (_Float16*)alloc((size_t)RS*KENC*sizeof(_Float16));
  p.Xg   = (_Float16*)alloc((size_t)RG*KGRD*sizeof(_Float16));
  p.Wenc = (_Float16*)alloc(512*KENC*sizeof(_Float16));
  p.Wdec = (_Float16*)alloc(512*KDEC*sizeof(_Float16));
  p.Wgrd = (_Float16*)alloc(512*KGRD*sizeof(_Float16));
  p.Weiw = (_Float16*)alloc(384*128*sizeof(_Float16));
  p.Weow = (_Float16*)alloc(128*128*sizeof(_Float16));
  p.Wdiw = (_Float16*)alloc(384*128*sizeof(_Float16));
  p.Wdow = (_Float16*)alloc(128*128*sizeof(_Float16));
  p.Benc = (float*)alloc(512*sizeof(float));
  p.Bdec = (float*)alloc(512*sizeof(float));
  p.Bgrd = (float*)alloc(512*sizeof(float));
  p.out  = (float*)d_out;

  const int prepTotal = 512*KENC + 512*KDEC + 512*KGRD + 2*(384*128) + 2*(128*128) + 3*512;
  prep_weights_kernel<<<(prepTotal+255)/256, 256, 0, stream>>>(p);
  const int preTotal = CB*13*24 + CS*16 + CG*16;
  precompute_kernel<<<(preTotal+255)/256, 256, 0, stream>>>(p);
  const int initTotal = RS*CH + RG*CH;
  init_state_kernel<<<(initTotal+255)/256, 256, 0, stream>>>(p);

  const int QT = NPAD / 16;

  for (int i = 0; i < CTIN; ++i) {
    build_stn_enc_kernel<<<(RS*KENC+255)/256, 256, 0, stream>>>(p, i);
    lstm_kernel<KENC><<<RS/16, 256, 0, stream>>>(p.Xs, p.Wenc, p.Benc, p.stn_h, p.stn_c);
    build_grid_enc_kernel<<<(RG*KGRD+255)/256, 256, 0, stream>>>(p, i);
    lstm_kernel<KGRD><<<RG/16, 256, 0, stream>>>(p.Xg, p.Wgrd, p.Bgrd, p.grid_h, p.grid_c);
    assemble_kernel<<<(CB*NPAD*CH+255)/256, 256, 0, stream>>>(p);
    qkv_kernel<<<RX/16, 256, 0, stream>>>(p.xh_h, p.Weiw, p.eib, p.qb, p.kb, p.vt);
    flash_kernel<<<CB*QT, 32, 0, stream>>>(p.qb, p.kb, p.vt, p.ctx, p.masks, i, 0);
    outproj_kernel<<<RX/16, 256, 0, stream>>>(p.ctx, p.Weow, p.eob, p.xh, p.xh_h);
    unpack_kernel<<<(CB*NTOK*CH+255)/256, 256, 0, stream>>>(p);
  }
  for (int i = 0; i < CTOUT; ++i) {
    build_stn_dec_kernel<<<(RS*KDEC+255)/256, 256, 0, stream>>>(p, i);
    lstm_kernel<KDEC><<<RS/16, 256, 0, stream>>>(p.Xs, p.Wdec, p.Bdec, p.stn_h, p.stn_c);
    assemble_kernel<<<(CB*NPAD*CH+255)/256, 256, 0, stream>>>(p);
    qkv_kernel<<<RX/16, 256, 0, stream>>>(p.xh_h, p.Wdiw, p.dib, p.qb, p.kb, p.vt);
    flash_kernel<<<CB*QT, 32, 0, stream>>>(p.qb, p.kb, p.vt, p.ctx, p.masks, i, 1);
    outproj_kernel<<<RX/16, 256, 0, stream>>>(p.ctx, p.Wdow, p.dob, p.xh, p.xh_h);
    unpack_kernel<<<(CB*NTOK*CH+255)/256, 256, 0, stream>>>(p);
    head_kernel<<<(RS+255)/256, 256, 0, stream>>>(p, i);
  }
}
